// CausalSelfAttention_71622874628697
// MI455X (gfx1250) — matrix operand order repaired and checked by reading, not compiled, not run on hardware
//
#include <hip/hip_runtime.h>

// ---------------------------------------------------------------------------
// CDNA5 (gfx1250) causal self-attention: QKV gemm -> flash attention -> proj
// All matmuls use v_wmma_f32_16x16x32_f16 (f16 in, f32 accumulate).
// Softmax row-sums are computed with a WMMA against an all-ones operand;
// softmax runs in the base-2 domain (v_exp_f32 is natively exp2).
// B=64, T=1024, C=64, H=4, D=16
// ---------------------------------------------------------------------------

typedef _Float16 v16h __attribute__((ext_vector_type(16)));
typedef _Float16 v8h  __attribute__((ext_vector_type(8)));
typedef float    v8f  __attribute__((ext_vector_type(8)));
typedef float    v4f  __attribute__((ext_vector_type(4)));

union V16 { v16h v; v8h h[2]; };

#define WMMA16(a, b, c) \
  __builtin_amdgcn_wmma_f32_16x16x32_f16(false, (a), false, (b), (short)0, (c), false, false)

__device__ __forceinline__ float redmax16(float v) {
#pragma unroll
  for (int m = 1; m < 16; m <<= 1) v = fmaxf(v, __shfl_xor(v, m, 32));
  return v;
}

// --------------------------- kernel 0: weights -> f16 ----------------------
__global__ void wcvt_kernel(const float* __restrict__ wq, const float* __restrict__ wp,
                            _Float16* __restrict__ wqh, _Float16* __restrict__ wph) {
  int i = blockIdx.x * blockDim.x + threadIdx.x;
  if (i < 192 * 64) wqh[i] = (_Float16)wq[i];
  if (i < 64 * 64)  wph[i] = (_Float16)wp[i];
}

// --------------------------- kernel 1: QKV projection ----------------------
// qkv[m, n] = sum_k x[m,k] * Wqkv[n,k];  M = B*T = 65536, N = 192, K = 64.
// Grid is nt-major: blockIdx.x = nt*512 + mb, so nt (and thus the q/k/v
// selector s = nt>>2 and head h = nt&3) is an SGPR -> scalar branches only.
__global__ void __launch_bounds__(256)
qkv_gemm_kernel(const float* __restrict__ x, const _Float16* __restrict__ wh,
                _Float16* __restrict__ Qh, _Float16* __restrict__ Kh,
                _Float16* __restrict__ Vt) {
  const int nt = blockIdx.x >> 9;          // 0..11  (block-uniform)
  const int mb = blockIdx.x & 511;         // group of 8 m-tiles
  const int wslot = threadIdx.x >> 5;
  const int lane = threadIdx.x & 31;
  const int g = lane >> 4;                 // lane half
  const int hn = lane & 15;                // A-row (m) / B-col (n) / D-col (n)
  const int mt = mb * 8 + wslot;
  const int gm0 = mt * 16;

  const float* xrow = x + (size_t)(gm0 + hn) * 64;
  v8f c = {};
#pragma unroll
  for (int k0 = 0; k0 < 64; k0 += 32) {
    // A: x[gm0+hn][k0 + e + 16g] converted to f16 (contiguous k order)
    V16 a;
    v4f x0 = *(const v4f*)(xrow + k0 + 16 * g);
    v4f x1 = *(const v4f*)(xrow + k0 + 16 * g + 4);
    v4f x2 = *(const v4f*)(xrow + k0 + 16 * g + 8);
    v4f x3 = *(const v4f*)(xrow + k0 + 16 * g + 12);
#pragma unroll
    for (int i = 0; i < 4; ++i) {
      a.v[i]      = (_Float16)x0[i];
      a.v[4 + i]  = (_Float16)x1[i];
      a.v[8 + i]  = (_Float16)x2[i];
      a.v[12 + i] = (_Float16)x3[i];
    }
    // B: Wqkv[nt*16+hn][k0 + e + 16g] (contiguous k order, same as A), contiguous 32B
    v16h b = *(const v16h*)(wh + (size_t)(nt * 16 + hn) * 64 + k0 + 16 * g);
    c = WMMA16(a.v, b, c);
  }

  // Scatter: D element (r, lane) is row gm0+r+8g, col hn of this (s,h) slice.
  const int s = nt >> 2, h = nt & 3;       // scalar (blockIdx-derived)
  const int b_ = gm0 >> 10;                // batch (tile never crosses batch)
  const int t0 = (gm0 & 1023) + 8 * g;
  const int bh = b_ * 4 + h;
  if (s == 0) {
    _Float16* dst = Qh + ((size_t)bh * 1024 + t0) * 16 + hn;
#pragma unroll
    for (int r = 0; r < 8; ++r) dst[r * 16] = (_Float16)c[r];
  } else if (s == 1) {
    _Float16* dst = Kh + ((size_t)bh * 1024 + t0) * 16 + hn;
#pragma unroll
    for (int r = 0; r < 8; ++r) dst[r * 16] = (_Float16)c[r];
  } else {
    // Transposed V: lane's 8 values are 8 consecutive t -> one b128 store.
    v8h pk;
#pragma unroll
    for (int r = 0; r < 8; ++r) pk[r] = (_Float16)c[r];
    *(v8h*)(Vt + ((size_t)bh * 16 + hn) * 1024 + t0) = pk;
  }
}

// --------------------------- kernel 2: flash attention ---------------------
// One wave per 16-row q-tile of one (b,h). Key tiles of 32.
// S = Q K^T via two WMMAs (K-dim padded 16->32); softmax row-sum via
// P @ ones WMMA; O += P V via one WMMA.  4 WMMAs per key tile.
__global__ void __launch_bounds__(256)
flash_kernel(const _Float16* __restrict__ Qh, const _Float16* __restrict__ Kh,
             const _Float16* __restrict__ Vt, _Float16* __restrict__ Y) {
  __shared__ __align__(128) _Float16 plds[8][16 * 32];   // per-wave P scratch (1KB)

  const int wslot = threadIdx.x >> 5;
  const int wave  = blockIdx.x * 8 + wslot;
  const int lane  = threadIdx.x & 31;
  const int g  = lane >> 4;
  const int hn = lane & 15;

  const int bh = wave >> 6;          // 64 q-tiles per (b,h)
  const int qt = wave & 63;
  const int q0 = qt * 16;

  const _Float16* Qb = Qh + (size_t)bh * 1024 * 16;
  const _Float16* Kb = Kh + (size_t)bh * 1024 * 16;
  const _Float16* Vb = Vt + (size_t)bh * 16 * 1024;
  _Float16* prow = &plds[wslot][0];

  // A operand for S: Q rows, K-dim 0..15 valid, 16..31 zero-padded.
  V16 aq;
  if (g == 0) aq.v = *(const v16h*)(Qb + (size_t)(q0 + hn) * 16);
  else {
#pragma unroll
  for (int i = 0; i < 16; ++i) aq.v[i] = (_Float16)0.0f;
  }

  // All-ones B operand: rowsum(P) = P @ ones, lands in C layout (no shuffles).
  V16 ones;
#pragma unroll
  for (int i = 0; i < 16; ++i) ones.v[i] = (_Float16)1.0f;

  float m_r[8], l_r[8], alpha_r[8];
#pragma unroll
  for (int r = 0; r < 8; ++r) { m_r[r] = -1e30f; l_r[r] = 0.0f; }
  v8f o = {};
  // Base-2 softmax: fold 1/sqrt(D) * log2(e) into the score scale.
  const float scale2 = 0.25f * 1.4426950408889634f;

  for (int k0 = 0; k0 <= q0 + 15; k0 += 32) {
    // B operands for S: lanes 0-15 carry K rows (all 16 d's), lanes 16-31 = pad.
    V16 bk0, bk1;
    if (g == 0) {
      bk0.v = *(const v16h*)(Kb + (size_t)(k0 + hn) * 16);
      bk1.v = *(const v16h*)(Kb + (size_t)(k0 + 16 + hn) * 16);
    } else {
#pragma unroll
      for (int i = 0; i < 16; ++i) { bk0.v[i] = (_Float16)0.0f; bk1.v[i] = (_Float16)0.0f; }
    }
    v8f zc = {};
    v8f c0 = WMMA16(aq.v, bk0.v, zc);   // keys k0 .. k0+15
    v8f c1 = WMMA16(aq.v, bk1.v, zc);   // keys k0+16 .. k0+31

    const bool need_mask = (k0 + 31 > q0);
#pragma unroll
    for (int r = 0; r < 8; ++r) {
      const int row = q0 + r + 8 * g;
      float s0 = c0[r] * scale2;
      float s1 = c1[r] * scale2;
      if (need_mask) {
        if (k0 + hn > row)      s0 = -1e30f;
        if (k0 + 16 + hn > row) s1 = -1e30f;
      }
      const float tmax = redmax16(fmaxf(s0, s1));
      const float mn = fmaxf(m_r[r], tmax);
      const float e0 = exp2f(s0 - mn);
      const float e1 = exp2f(s1 - mn);
      const float alpha = exp2f(m_r[r] - mn);
      alpha_r[r] = alpha;
      m_r[r] = mn;
      o[r] *= alpha;
      // P into LDS in row-major [16 rows][32 keys]
      prow[(r + 8 * g) * 32 + hn]      = (_Float16)e0;
      prow[(r + 8 * g) * 32 + 16 + hn] = (_Float16)e1;
    }

    // Reload P in WMMA A layout (same-wave LDS ops are in order).
    V16 pa;
    pa.h[0] = *(const v8h*)(prow + hn * 32 + 16 * g);
    pa.h[1] = *(const v8h*)(prow + hn * 32 + 16 * g + 8);

    // Row sums of P via WMMA: cs[r] = sum_k P[r+8g][k] (replicated over lanes).
    v8f cs = WMMA16(pa.v, ones.v, zc);
#pragma unroll
    for (int r = 0; r < 8; ++r) l_r[r] = l_r[r] * alpha_r[r] + cs[r];

    // V B-operand: element e -> V[k0 + e + 16g][hn] = Vt[hn][k0+16g+e], contiguous.
    v16h bv = *(const v16h*)(Vb + (size_t)hn * 1024 + k0 + 16 * g);
    o = WMMA16(pa.v, bv, o);
  }

  // Normalize and emit y[b, t, h*16 + d] as f16 for the projection GEMM.
  const int b_ = bh >> 2, h = bh & 3;
  _Float16* ydst = Y + ((size_t)(b_ * 1024 + q0 + 8 * g)) * 64 + h * 16 + hn;
#pragma unroll
  for (int r = 0; r < 8; ++r)
    ydst[(size_t)r * 64] = (_Float16)(o[r] / l_r[r]);
}

// --------------------------- kernel 3: output projection -------------------
// out[m, n] = sum_k Y[m,k] * Wproj[n,k];  M = 65536, N = 64, K = 64.
__global__ void __launch_bounds__(256)
proj_kernel(const _Float16* __restrict__ Y, const _Float16* __restrict__ wph,
            float* __restrict__ out) {
  const int wave = blockIdx.x * (blockDim.x >> 5) + (threadIdx.x >> 5);
  const int lane = threadIdx.x & 31;
  const int g = lane >> 4;
  const int hn = lane & 15;
  const int mt = wave >> 2;
  const int nt = wave & 3;
  const int gm0 = mt * 16;

  const _Float16* yrow = Y + (size_t)(gm0 + hn) * 64;
  v8f c = {};
#pragma unroll
  for (int k0 = 0; k0 < 64; k0 += 32) {
    V16 a;
    a.h[0] = *(const v8h*)(yrow + k0 + 16 * g);
    a.h[1] = *(const v8h*)(yrow + k0 + 16 * g + 8);
    v16h b = *(const v16h*)(wph + (size_t)(nt * 16 + hn) * 64 + k0 + 16 * g);
    c = WMMA16(a.v, b, c);
  }
  float* dst = out + (size_t)(gm0 + 8 * g) * 64 + nt * 16 + hn;
#pragma unroll
  for (int r = 0; r < 8; ++r)
    dst[(size_t)r * 64] = c[r];
}

// ---------------------------------------------------------------------------
extern "C" void kernel_launch(void* const* d_in, const int* in_sizes, int n_in,
                              void* d_out, int out_size, void* d_ws, size_t ws_size,
                              hipStream_t stream) {
  (void)in_sizes; (void)n_in; (void)out_size; (void)ws_size;
  const float* x  = (const float*)d_in[0];   // (64,1024,64)
  const float* wq = (const float*)d_in[1];   // (192,64)
  const float* wp = (const float*)d_in[2];   // (64,64)
  float* out = (float*)d_out;                // (64,1024,64)

  char* ws = (char*)d_ws;
  const size_t MB8 = 8ull << 20;
  _Float16* Qh  = (_Float16*)(ws);             // [B*H][T][16]   8 MB
  _Float16* Kh  = (_Float16*)(ws + MB8);       // [B*H][T][16]   8 MB
  _Float16* Vt  = (_Float16*)(ws + 2 * MB8);   // [B*H][16][T]   8 MB (transposed)
  _Float16* Yh  = (_Float16*)(ws + 3 * MB8);   // [B*T][64]      8 MB
  _Float16* wqh = (_Float16*)(ws + 4 * MB8);   // 24 KB
  _Float16* wph = (_Float16*)(ws + 4 * MB8 + 32768); // 8 KB

  wcvt_kernel<<<48, 256, 0, stream>>>(wq, wp, wqh, wph);
  // nt-major grid: 12 n-tiles * 512 blocks (each block = 8 m-tiles)
  qkv_gemm_kernel<<<12 * 512, 256, 0, stream>>>(x, wqh, Qh, Kh, Vt);
  // 64*4*(1024/16) = 16384 q-tile waves / 8 per block
  flash_kernel<<<2048, 256, 0, stream>>>(Qh, Kh, Vt, Yh);
  // 4096 m-tiles * 4 n-tiles = 16384 waves / 8 per block
  proj_kernel<<<2048, 256, 0, stream>>>(Yh, wph, out);
}
